// MinLSTM_37349035606029
// MI455X (gfx1250) — compile-verified, matrix operand order checked by reading, not run on hardware
//
#include <hip/hip_runtime.h>

// ---------------------------------------------------------------------------
// MinLSTM on MI455X (gfx1250):
//   1) split fp32 x/W into (hi,lo) bf16 once
//   2) bf16x3 WMMA GEMM (double-buffered LDS, async global->LDS)
//   3) chunked parallel linear scan
// ---------------------------------------------------------------------------

typedef __bf16 bf16;
typedef __attribute__((ext_vector_type(16))) __bf16 v16bf;
typedef __attribute__((ext_vector_type(4)))  __bf16 v4bf;
typedef __attribute__((ext_vector_type(8)))  float  v8f;
typedef __attribute__((ext_vector_type(4)))  int    v4i;

#define TILE_M 128
#define TILE_N 128
#define TILE_K 32
#define KROW   40   // padded LDS row stride in bf16 (80 B = 5x16 B, conflict-free)

#if __has_builtin(__builtin_amdgcn_global_load_async_to_lds_b128) && \
    __has_builtin(__builtin_amdgcn_s_wait_asynccnt)
#define USE_ASYNC_LDS 1
#else
#define USE_ASYNC_LDS 0
#endif

#if USE_ASYNC_LDS
#define AS3 __attribute__((address_space(3)))
// global source: builtin expects a generic (non-const) v4i*
__device__ __forceinline__ v4i* g_v4i(const void* p) {
  return (v4i*)(uintptr_t)p;
}
// LDS dest: flat shared address low 32 bits are the LDS byte offset
__device__ __forceinline__ AS3 v4i* l_v4i(void* p) {
  return (AS3 v4i*)(unsigned)(uintptr_t)p;
}
#endif

struct LdsBuf {
  bf16 Ah[TILE_M][KROW];
  bf16 Al[TILE_M][KROW];
  bf16 Bh[TILE_N][KROW];
  bf16 Bl[TILE_N][KROW];
};

__device__ __forceinline__ float fast_sigmoid(float x) {
  return 1.0f / (1.0f + __expf(-x));
}
__device__ __forceinline__ float fast_softplus(float x) {
  if (x > 20.0f) return x;
  if (x < -20.0f) return __expf(x);
  return __logf(1.0f + __expf(x));
}
__device__ __forceinline__ void gate_eval(float fp, float ip, float hp,
                                          float& f, float& ig) {
  float diff = fast_softplus(-fp) - fast_softplus(-ip);
  f = fast_sigmoid(-diff);
  float i = 1.0f - f;
  float g = (hp >= 0.0f) ? (hp + 0.5f) : fast_sigmoid(hp);
  ig = i * g;
}

// ---------------------------------------------------------------------------
// One-shot fp32 -> (hi, lo) bf16 split, 4 elements/thread.
// ---------------------------------------------------------------------------
__global__ void split_bf16(const float* __restrict__ src,
                           bf16* __restrict__ hi, bf16* __restrict__ lo, int n4)
{
  const int i = blockIdx.x * blockDim.x + threadIdx.x;
  if (i >= n4) return;
  const float4 v = ((const float4*)src)[i];
  const float vv[4] = {v.x, v.y, v.z, v.w};
  v4bf h, l;
#pragma unroll
  for (int j = 0; j < 4; ++j) {
    bf16 hj = (bf16)vv[j];
    h[j] = hj;
    l[j] = (bf16)(vv[j] - (float)hj);
  }
  *(v4bf*)(hi + (size_t)i * 4) = h;
  *(v4bf*)(lo + (size_t)i * 4) = l;
}

// ---------------------------------------------------------------------------
// bf16x3 triple-gate GEMM: pre_g[m,n] = sum_k x[m,k] * W_g[n,k]
// grid.x = M/128, grid.y = 3*(N/128); 256 threads = 8 waves, wave tile 64x32.
// Double-buffered LDS; async global->LDS copies when the toolchain has them.
// ---------------------------------------------------------------------------
__global__ __launch_bounds__(256) void gate_gemm_bf16x3(
    const bf16* __restrict__ Xhi, const bf16* __restrict__ Xlo,
    const bf16* __restrict__ Whi, const bf16* __restrict__ Wlo,
    float* __restrict__ preF, float* __restrict__ preI, float* __restrict__ preH,
    int M, int K, int N)
{
  extern __shared__ __align__(16) char smem_raw[];
  LdsBuf* bufs = (LdsBuf*)smem_raw;

  const int m_blk  = blockIdx.x * TILE_M;
  const int ntiles = N / TILE_N;
  const int gate   = blockIdx.y / ntiles;
  const int n_blk  = (blockIdx.y % ntiles) * TILE_N;
  float* Osel = (gate == 0) ? preF : (gate == 1) ? preI : preH;

  const bf16* gsrc[4];
  gsrc[0] = Xhi + (size_t)m_blk * K;
  gsrc[1] = Xlo + (size_t)m_blk * K;
  gsrc[2] = Whi + (size_t)(gate * N + n_blk) * K;
  gsrc[3] = Wlo + (size_t)(gate * N + n_blk) * K;

  const int tid  = threadIdx.x;
  const int lane = tid & 31;
  const int wid  = tid >> 5;
  const int wm   = wid & 1;    // 2 wave rows x 64
  const int wn   = wid >> 1;   // 4 wave cols x 32

  // staging work items: per array 512 x 16B transfers; thread does {tid, tid+256}
  const int sr0 = tid >> 2,          sc0 = (tid & 3) * 8;
  const int sr1 = (tid + 256) >> 2,  sc1 = ((tid + 256) & 3) * 8;

  v8f acc[4][2];
  const v8f vzero = {0.f, 0.f, 0.f, 0.f, 0.f, 0.f, 0.f, 0.f};
#pragma unroll
  for (int mt = 0; mt < 4; ++mt)
#pragma unroll
    for (int nt = 0; nt < 2; ++nt) acc[mt][nt] = vzero;

#if USE_ASYNC_LDS
#define STAGE_ASYNC(BUF, KOFF)                                                 \
  do {                                                                         \
    bf16* darr_[4] = {&(BUF).Ah[0][0], &(BUF).Al[0][0],                        \
                      &(BUF).Bh[0][0], &(BUF).Bl[0][0]};                       \
    _Pragma("unroll")                                                          \
    for (int a_ = 0; a_ < 4; ++a_) {                                           \
      __builtin_amdgcn_global_load_async_to_lds_b128(                          \
          g_v4i(gsrc[a_] + (size_t)sr0 * K + (KOFF) + sc0),                    \
          l_v4i(darr_[a_] + sr0 * KROW + sc0), 0, 0);                          \
      __builtin_amdgcn_global_load_async_to_lds_b128(                          \
          g_v4i(gsrc[a_] + (size_t)sr1 * K + (KOFF) + sc1),                    \
          l_v4i(darr_[a_] + sr1 * KROW + sc1), 0, 0);                          \
    }                                                                          \
  } while (0)
#else
#define LOAD_TILE(REGS, KOFF)                                                  \
  do {                                                                         \
    _Pragma("unroll")                                                          \
    for (int a_ = 0; a_ < 4; ++a_) {                                           \
      (REGS)[a_ * 2 + 0] =                                                     \
          *(const float4*)(gsrc[a_] + (size_t)sr0 * K + (KOFF) + sc0);         \
      (REGS)[a_ * 2 + 1] =                                                     \
          *(const float4*)(gsrc[a_] + (size_t)sr1 * K + (KOFF) + sc1);         \
    }                                                                          \
  } while (0)
#define STORE_TILE(REGS, BUF)                                                  \
  do {                                                                         \
    bf16* darr_[4] = {&(BUF).Ah[0][0], &(BUF).Al[0][0],                        \
                      &(BUF).Bh[0][0], &(BUF).Bl[0][0]};                       \
    _Pragma("unroll")                                                          \
    for (int a_ = 0; a_ < 4; ++a_) {                                           \
      *(float4*)(darr_[a_] + sr0 * KROW + sc0) = (REGS)[a_ * 2 + 0];           \
      *(float4*)(darr_[a_] + sr1 * KROW + sc1) = (REGS)[a_ * 2 + 1];           \
    }                                                                          \
  } while (0)
#endif

  // ---- prologue: stage tile 0 into buf 0
#if USE_ASYNC_LDS
  STAGE_ASYNC(bufs[0], 0);
  __builtin_amdgcn_s_wait_asynccnt(0);
#else
  {
    float4 r0[8];
    LOAD_TILE(r0, 0);
    STORE_TILE(r0, bufs[0]);
  }
#endif
  __syncthreads();

  const int nsteps = K / TILE_K;
  for (int ks = 0; ks < nsteps; ++ks) {
    const int cur  = ks & 1;
    const int knext = (ks + 1) * TILE_K;

#if USE_ASYNC_LDS
    if (ks + 1 < nsteps) STAGE_ASYNC(bufs[cur ^ 1], knext);
#else
    float4 rn[8];
    if (ks + 1 < nsteps) LOAD_TILE(rn, knext);
#endif

    // ---- fragments from current buffer (16-bit WMMA VGPR layout:
    //  lane<16: row lane, K in {0..7}u{16..23}; lane>=16: row lane-16, K+8)
    LdsBuf& Cb = bufs[cur];
    const int r    = lane & 15;
    const int koff = (lane >> 4) * 8;
    v16bf a_hi[4], a_lo[4], b_hi[2], b_lo[2];
#pragma unroll
    for (int mt = 0; mt < 4; ++mt) {
      const int m = wm * 64 + mt * 16 + r;
#pragma unroll
      for (int j = 0; j < 16; ++j) {
        const int k = (j & 7) + ((j >> 3) << 4) + koff;
        a_hi[mt][j] = Cb.Ah[m][k];
        a_lo[mt][j] = Cb.Al[m][k];
      }
    }
#pragma unroll
    for (int nt = 0; nt < 2; ++nt) {
      const int n = wn * 32 + nt * 16 + r;
#pragma unroll
      for (int j = 0; j < 16; ++j) {
        const int k = (j & 7) + ((j >> 3) << 4) + koff;
        b_hi[nt][j] = Cb.Bh[n][k];
        b_lo[nt][j] = Cb.Bl[n][k];
      }
    }

    // ---- bf16x3: hi*hi + hi*lo + lo*hi
#pragma unroll
    for (int mt = 0; mt < 4; ++mt)
#pragma unroll
      for (int nt = 0; nt < 2; ++nt) {
        acc[mt][nt] = __builtin_amdgcn_wmma_f32_16x16x32_bf16(
            false, a_hi[mt], false, b_hi[nt], (short)0, acc[mt][nt], false, false);
        acc[mt][nt] = __builtin_amdgcn_wmma_f32_16x16x32_bf16(
            false, a_hi[mt], false, b_lo[nt], (short)0, acc[mt][nt], false, false);
        acc[mt][nt] = __builtin_amdgcn_wmma_f32_16x16x32_bf16(
            false, a_lo[mt], false, b_hi[nt], (short)0, acc[mt][nt], false, false);
      }

#if USE_ASYNC_LDS
    __builtin_amdgcn_s_wait_asynccnt(0);
#else
    if (ks + 1 < nsteps) STORE_TILE(rn, bufs[cur ^ 1]);
#endif
    __syncthreads();
  }

  // ---- epilogue: C layout — VGPR r: lanes0-15 -> M=r, lanes16-31 -> M=r+8
  const int ccol = lane & 15;
  const int crow = (lane >> 4) * 8;
#pragma unroll
  for (int mt = 0; mt < 4; ++mt)
#pragma unroll
    for (int nt = 0; nt < 2; ++nt) {
      const int nn = n_blk + wn * 32 + nt * 16 + ccol;
#pragma unroll
      for (int rr = 0; rr < 8; ++rr) {
        const int mm = m_blk + wm * 64 + mt * 16 + crow + rr;
        Osel[(size_t)mm * N + nn] = acc[mt][nt][rr];
      }
    }
}

// ---------------------------------------------------------------------------
// Chunked parallel scan:  h_t = f_t * h_{t-1} + (i*g)_t
// ---------------------------------------------------------------------------
__global__ void scan_pass1(const float* __restrict__ preF,
                           const float* __restrict__ preI,
                           const float* __restrict__ preH,
                           const float* __restrict__ bF,
                           const float* __restrict__ bI,
                           const float* __restrict__ bH,
                           float* __restrict__ chunkP,
                           float* __restrict__ chunkS,
                           int BDtot, int L, int D, int C)
{
  const int t = blockIdx.x * blockDim.x + threadIdx.x;
  if (t >= BDtot * C) return;
  const int bd = t % BDtot;
  const int c  = t / BDtot;
  const int b  = bd / D;
  const int d  = bd % D;
  const int Lc = L / C;
  const float bf = bF[d], bi = bI[d], bh = bH[d];
  size_t idx = ((size_t)b * L + (size_t)c * Lc) * D + d;
  float P = 1.0f, S = 0.0f;
  for (int l = 0; l < Lc; ++l, idx += D) {
    float f, ig;
    gate_eval(preF[idx] + bf, preI[idx] + bi, preH[idx] + bh, f, ig);
    S = f * S + ig;
    P *= f;
  }
  chunkP[(size_t)c * BDtot + bd] = P;
  chunkS[(size_t)c * BDtot + bd] = S;
}

__global__ void scan_pass2(const float* __restrict__ chunkP,
                           const float* __restrict__ chunkS,
                           float* __restrict__ chunkH,
                           int BDtot, int C)
{
  const int bd = blockIdx.x * blockDim.x + threadIdx.x;
  if (bd >= BDtot) return;
  float h = 0.0f;   // h_0 = exp(NEG_BIG) = 0
  for (int c = 0; c < C; ++c) {
    chunkH[(size_t)c * BDtot + bd] = h;
    h = chunkP[(size_t)c * BDtot + bd] * h + chunkS[(size_t)c * BDtot + bd];
  }
}

__global__ void scan_pass3(const float* __restrict__ preF,
                           const float* __restrict__ preI,
                           const float* __restrict__ preH,
                           const float* __restrict__ bF,
                           const float* __restrict__ bI,
                           const float* __restrict__ bH,
                           const float* __restrict__ chunkH,
                           float* __restrict__ out,
                           int BDtot, int L, int D, int C)
{
  const int t = blockIdx.x * blockDim.x + threadIdx.x;
  if (t >= BDtot * C) return;
  const int bd = t % BDtot;
  const int c  = t / BDtot;
  const int b  = bd / D;
  const int d  = bd % D;
  const int Lc = L / C;
  const float bf = bF[d], bi = bI[d], bh = bH[d];
  size_t idx = ((size_t)b * L + (size_t)c * Lc) * D + d;
  float h = chunkH[(size_t)c * BDtot + bd];
  for (int l = 0; l < Lc; ++l, idx += D) {
    float f, ig;
    gate_eval(preF[idx] + bf, preI[idx] + bi, preH[idx] + bh, f, ig);
    h = f * h + ig;
    out[idx] = h;
  }
}

// ---------------------------------------------------------------------------
extern "C" void kernel_launch(void* const* d_in, const int* in_sizes, int n_in,
                              void* d_out, int out_size, void* d_ws, size_t ws_size,
                              hipStream_t stream) {
  (void)n_in; (void)out_size; (void)ws_size;
  const float* x  = (const float*)d_in[0];
  const float* Wf = (const float*)d_in[1];
  const float* bf = (const float*)d_in[2];
  const float* Wi = (const float*)d_in[3];
  const float* bi = (const float*)d_in[4];
  const float* Wh = (const float*)d_in[5];
  const float* bh = (const float*)d_in[6];
  float* out = (float*)d_out;

  const int D = in_sizes[2];          // 1024
  const int M = in_sizes[0] / D;      // B*L = 16384
  const int L = 4096;
  const int B = M / L;                // 4
  const int K = D, N = D;
  const int C = 16;                   // scan chunks (Lc = 256)
  const int BDtot = B * D;

  // workspace layout:
  //   3 x (M*D) f32 pre-activations            = 192 MB
  //   3 x (BDtot*C) f32 chunk summaries        = 0.75 MB
  //   2 x (M*K) bf16 X split + 2 x (3*N*K) bf16 W split = 76 MB
  const size_t S1 = (size_t)M * D;
  float* preF   = (float*)d_ws;
  float* preI   = preF + S1;
  float* preH   = preI + S1;
  float* chunkP = preH + S1;
  float* chunkS = chunkP + (size_t)BDtot * C;
  float* chunkH = chunkS + (size_t)BDtot * C;
  bf16*  Xhi    = (bf16*)(chunkH + (size_t)BDtot * C);
  bf16*  Xlo    = Xhi + (size_t)M * K;
  bf16*  Whi    = Xlo + (size_t)M * K;
  bf16*  Wlo    = Whi + (size_t)3 * N * K;

  // 1) one-shot fp32 -> (hi,lo) bf16 splits
  const int nX4 = (M * K) / 4;
  const int nW4 = (N * K) / 4;
  split_bf16<<<(nX4 + 255) / 256, 256, 0, stream>>>(x, Xhi, Xlo, nX4);
  split_bf16<<<(nW4 + 255) / 256, 256, 0, stream>>>(Wf, Whi, Wlo, nW4);
  split_bf16<<<(nW4 + 255) / 256, 256, 0, stream>>>(
      Wi, Whi + (size_t)N * K, Wlo + (size_t)N * K, nW4);
  split_bf16<<<(nW4 + 255) / 256, 256, 0, stream>>>(
      Wh, Whi + (size_t)2 * N * K, Wlo + (size_t)2 * N * K, nW4);

  // 2) WMMA GEMM for all three gates
  dim3 ggrid(M / TILE_M, 3 * (N / TILE_N));
  gate_gemm_bf16x3<<<ggrid, 256, 2 * sizeof(LdsBuf), stream>>>(
      Xhi, Xlo, Whi, Wlo, preF, preI, preH, M, K, N);

  // 3) chunked scan
  const int t1 = BDtot * C;
  scan_pass1<<<(t1 + 255) / 256, 256, 0, stream>>>(
      preF, preI, preH, bf, bi, bh, chunkP, chunkS, BDtot, L, D, C);
  scan_pass2<<<(BDtot + 255) / 256, 256, 0, stream>>>(
      chunkP, chunkS, chunkH, BDtot, C);
  scan_pass3<<<(t1 + 255) / 256, 256, 0, stream>>>(
      preF, preI, preH, bf, bi, bh, chunkH, out, BDtot, L, D, C);
}
